// MLP_40097814675501
// MI455X (gfx1250) — matrix operand order repaired and checked by reading, not compiled, not run on hardware
//
#include <hip/hip_runtime.h>

// ---------------------------------------------------------------------------
// Types
// ---------------------------------------------------------------------------
typedef __bf16 bf16;
typedef __bf16 v16bf __attribute__((ext_vector_type(16)));
typedef __bf16 v8bf  __attribute__((ext_vector_type(8)));
typedef __bf16 v4bf  __attribute__((ext_vector_type(4)));
typedef float  v8f   __attribute__((ext_vector_type(8)));

// Problem constants (B=4, S=2048, D=2048, H=4*D=8192)
#define MM   8192          // B*S rows
#define DD   2048
#define HH   8192
#define WCNT 16777216      // 8192*2048 elements in each weight matrix

// Workspace layout (bytes)
#define OFF_A1     ((size_t)0)                       // 8192*2048 bf16 = 32MB
#define OFF_WFCQ   ((size_t)33554432)                // 32MB
#define OFF_WPRQ   ((size_t)67108864)                // 32MB
#define OFF_H      ((size_t)100663296)               // 8192*8192 bf16 = 128MB
#define OFF_PARTA  ((size_t)234881024)               // 2048 floats
#define OFF_PARTB  ((size_t)234889216)               // 2048 floats
#define OFF_GAMMA  ((size_t)234897408)               // 2 floats

// ---------------------------------------------------------------------------
// CDNA5 async copy: 16 bytes / lane, global -> LDS, tracked by ASYNCcnt.
// ---------------------------------------------------------------------------
__device__ __forceinline__ void async_copy_b128(const bf16* gsrc, bf16* ldst) {
  unsigned lofs =
      (unsigned)(unsigned long long)(__attribute__((address_space(3))) bf16*)ldst;
  asm volatile("global_load_async_to_lds_b128 %0, %1, off"
               :
               : "v"(lofs), "v"((unsigned long long)gsrc)
               : "memory");
}
__device__ __forceinline__ void wait_async_le4() {
  asm volatile("s_wait_asynccnt 0x4" ::: "memory");
}
__device__ __forceinline__ void wait_async_0() {
  asm volatile("s_wait_asynccnt 0x0" ::: "memory");
}

// ---------------------------------------------------------------------------
// Block-wide reductions (blockDim.x == 256, wave32 -> 8 waves). Deterministic.
// ---------------------------------------------------------------------------
__device__ __forceinline__ float block_sum256(float v, float* sm) {
  const int lane = threadIdx.x & 31;
  const int wid  = threadIdx.x >> 5;
#pragma unroll
  for (int o = 16; o > 0; o >>= 1) v += __shfl_down(v, o, 32);
  __syncthreads();
  if (lane == 0) sm[wid] = v;
  __syncthreads();
  float s = 0.f;
#pragma unroll
  for (int i = 0; i < 8; ++i) s += sm[i];
  __syncthreads();
  return s;
}

__device__ __forceinline__ float block_max256(float v, float* sm) {
  const int lane = threadIdx.x & 31;
  const int wid  = threadIdx.x >> 5;
#pragma unroll
  for (int o = 16; o > 0; o >>= 1) v = fmaxf(v, __shfl_down(v, o, 32));
  __syncthreads();
  if (lane == 0) sm[wid] = v;
  __syncthreads();
  float s = sm[0];
#pragma unroll
  for (int i = 1; i < 8; ++i) s = fmaxf(s, sm[i]);
  __syncthreads();
  return s;
}

// ---------------------------------------------------------------------------
// gamma_w = mean(|w|): two-stage deterministic reduction
// ---------------------------------------------------------------------------
__global__ __launch_bounds__(256) void absmean_stage1(const float* __restrict__ w,
                                                      float* __restrict__ part) {
  __shared__ float sm[8];
  const size_t base = (size_t)blockIdx.x * (256 * 32);
  float s = 0.f;
#pragma unroll
  for (int i = 0; i < 32; ++i) s += fabsf(w[base + threadIdx.x + (size_t)i * 256]);
  s = block_sum256(s, sm);
  if (threadIdx.x == 0) part[blockIdx.x] = s;
}

__global__ __launch_bounds__(256) void absmean_stage2(const float* __restrict__ part,
                                                      float* __restrict__ gamma_out,
                                                      float inv_count) {
  __shared__ float sm[8];
  float s = 0.f;
#pragma unroll
  for (int i = 0; i < 8; ++i) s += part[threadIdx.x + i * 256];
  s = block_sum256(s, sm);
  if (threadIdx.x == 0) *gamma_out = s * inv_count;
}

// ---------------------------------------------------------------------------
// Ternarize: wq = sign(rint(w/(gamma+eps))) in {-1,0,1}, stored as bf16 (exact)
// ---------------------------------------------------------------------------
__global__ __launch_bounds__(256) void ternarize_k(const float* __restrict__ w,
                                                   const float* __restrict__ gammaPtr,
                                                   bf16* __restrict__ out, size_t n) {
  const float g = *gammaPtr + 1e-6f;
  const size_t i = ((size_t)blockIdx.x * 256 + threadIdx.x) * 4;
  if (i >= n) return;
  const float4 v = *(const float4*)(w + i);
  float4 r;
  r.x = rintf(v.x / g); r.y = rintf(v.y / g);
  r.z = rintf(v.z / g); r.w = rintf(v.w / g);
  v4bf o;
  o[0] = (bf16)((r.x > 0.f) ? 1.f : (r.x < 0.f ? -1.f : 0.f));
  o[1] = (bf16)((r.y > 0.f) ? 1.f : (r.y < 0.f ? -1.f : 0.f));
  o[2] = (bf16)((r.z > 0.f) ? 1.f : (r.z < 0.f ? -1.f : 0.f));
  o[3] = (bf16)((r.w > 0.f) ? 1.f : (r.w < 0.f ? -1.f : 0.f));
  *(v4bf*)(out + i) = o;
}

// ---------------------------------------------------------------------------
// LayerNorm + absmax clip ("fake quant", no rounding) -> bf16.  D = 2048.
// ---------------------------------------------------------------------------
__global__ __launch_bounds__(256) void ln_quant_x_k(const float* __restrict__ x,
                                                    bf16* __restrict__ out) {
  __shared__ float sm[8];
  const size_t row = blockIdx.x;
  const float* xr = x + row * (size_t)DD;
  const int base = threadIdx.x * 8;
  float v[8];
  const float4 p0 = *(const float4*)(xr + base);
  const float4 p1 = *(const float4*)(xr + base + 4);
  v[0] = p0.x; v[1] = p0.y; v[2] = p0.z; v[3] = p0.w;
  v[4] = p1.x; v[5] = p1.y; v[6] = p1.z; v[7] = p1.w;
  float s = 0.f, s2 = 0.f;
#pragma unroll
  for (int i = 0; i < 8; ++i) { s += v[i]; s2 += v[i] * v[i]; }
  s  = block_sum256(s, sm);
  s2 = block_sum256(s2, sm);
  const float mean = s * (1.0f / DD);
  const float var  = s2 * (1.0f / DD) - mean * mean;
  const float rstd = rsqrtf(var + 1e-6f);
  float amax = 0.f;
#pragma unroll
  for (int i = 0; i < 8; ++i) { v[i] = (v[i] - mean) * rstd; amax = fmaxf(amax, fabsf(v[i])); }
  const float g = block_max256(amax, sm);
  const float sIn = 128.0f / g, sOut = g * (1.0f / 128.0f);
  const float lo = -128.0f + 1e-6f, hi = 128.0f - 1e-6f;
  v8bf o;
#pragma unroll
  for (int i = 0; i < 8; ++i)
    o[i] = (bf16)(fminf(fmaxf(v[i] * sIn, lo), hi) * sOut);
  *(v8bf*)(out + row * (size_t)DD + base) = o;
}

// Same, over the GELU output (bf16, in place). D = 8192, 32 elements/thread.
__global__ __launch_bounds__(256) void ln_quant_h_k(bf16* __restrict__ h) {
  __shared__ float sm[8];
  const size_t row = blockIdx.x;
  bf16* hr = h + row * (size_t)HH;
  const int base = threadIdx.x * 32;
  float v[32];
#pragma unroll
  for (int c = 0; c < 4; ++c) {
    const v8bf t = *(const v8bf*)(hr + base + c * 8);
#pragma unroll
    for (int j = 0; j < 8; ++j) v[c * 8 + j] = (float)t[j];
  }
  float s = 0.f, s2 = 0.f;
#pragma unroll
  for (int i = 0; i < 32; ++i) { s += v[i]; s2 += v[i] * v[i]; }
  s  = block_sum256(s, sm);
  s2 = block_sum256(s2, sm);
  const float mean = s * (1.0f / HH);
  const float var  = s2 * (1.0f / HH) - mean * mean;
  const float rstd = rsqrtf(var + 1e-6f);
  float amax = 0.f;
#pragma unroll
  for (int i = 0; i < 32; ++i) { v[i] = (v[i] - mean) * rstd; amax = fmaxf(amax, fabsf(v[i])); }
  const float g = block_max256(amax, sm);
  const float sIn = 128.0f / g, sOut = g * (1.0f / 128.0f);
  const float lo = -128.0f + 1e-6f, hi = 128.0f - 1e-6f;
#pragma unroll
  for (int c = 0; c < 4; ++c) {
    v8bf o;
#pragma unroll
    for (int j = 0; j < 8; ++j)
      o[j] = (bf16)(fminf(fmaxf(v[c * 8 + j] * sIn, lo), hi) * sOut);
    *(v8bf*)(hr + base + c * 8) = o;
  }
}

// ---------------------------------------------------------------------------
// WMMA GEMM:  C[m,n] = gamma_w * sum_k A[m,k] * Wq[n,k]
//   A : (M x K) bf16 row-major; Wq: (N x K) bf16 row-major ({-1,0,1})
// Block: 256 threads = 8 waves; block tile 128(M) x 128(N); K-step 32.
// Wave (wm 0..3, wn 0..1) owns 32(M) x 64(N) = 2x4 WMMA accumulators.
// 3-deep LDS pipeline fed by global_load_async_to_lds_b128 (ASYNCcnt):
// one s_wait_asynccnt + one barrier per K-step.
// ---------------------------------------------------------------------------
#define LDST 40                  // padded LDS row stride (elements)
#define TBUF (128 * LDST)        // one A or B buffer, elements (10240 B)

template <bool GELU_BF16>
__global__ __launch_bounds__(256) void gemm_wmma_k(const bf16* __restrict__ A,
                                                   const bf16* __restrict__ W,
                                                   const float* __restrict__ gammaPtr,
                                                   void* __restrict__ Out,
                                                   int M, int N, int K) {
  __shared__ bf16 As[3][TBUF];   // 3 x 10240 B
  __shared__ bf16 Bs[3][TBUF];   // 3 x 10240 B   (total 61440 B)

  const int tid  = threadIdx.x;
  const int lane = tid & 31;
  const int wid  = tid >> 5;
  const int wm   = wid & 3;        // M sub-block (x32 rows)
  const int wn   = wid >> 2;       // N sub-block (x64 cols)
  const int mBase = blockIdx.y * 128;
  const int nBase = blockIdx.x * 128;

  // --- staging addresses: A/B each 128x32 = 512 x 16B chunks, 2 per thread ---
  const int srow0 = tid >> 2;            // 0..63
  const int srow1 = 64 + srow0;          // 64..127
  const int skc   = (tid & 3) * 8;       // 0,8,16,24
  const bf16* aS0 = A + (size_t)(mBase + srow0) * K + skc;
  const bf16* aS1 = A + (size_t)(mBase + srow1) * K + skc;
  const bf16* bS0 = W + (size_t)(nBase + srow0) * K + skc;
  const bf16* bS1 = W + (size_t)(nBase + srow1) * K + skc;
  const int dOfs0 = srow0 * LDST + skc;
  const int dOfs1 = srow1 * LDST + skc;

  auto issueTile = [&](int kOff, int buf) {
    async_copy_b128(aS0 + kOff, &As[buf][dOfs0]);
    async_copy_b128(aS1 + kOff, &As[buf][dOfs1]);
    async_copy_b128(bS0 + kOff, &Bs[buf][dOfs0]);
    async_copy_b128(bS1 + kOff, &Bs[buf][dOfs1]);
  };

  // --- fragment lane geometry (ISA layouts) ---
  // A (16x32): lane m holds K in [h*8,h*8+8) U [16+h*8,16+h*8+8)
  // B (32x16): lane n holds K in [h*8,h*8+8) U [16+h*8,16+h*8+8)
  const int lrow = lane & 15;
  const int half = lane >> 4;
  const int aOff = half * 8;
  const int bOff = half * 8;

  const v8f vzero = {0.f, 0.f, 0.f, 0.f, 0.f, 0.f, 0.f, 0.f};
  v8f acc[2][4];
#pragma unroll
  for (int s2 = 0; s2 < 2; ++s2)
#pragma unroll
    for (int t2 = 0; t2 < 4; ++t2) acc[s2][t2] = vzero;

  auto computeTile = [&](int buf) {
    v16bf aF[2], bF[4];
#pragma unroll
    for (int s2 = 0; s2 < 2; ++s2) {
      const bf16* p = &As[buf][(wm * 32 + s2 * 16 + lrow) * LDST];
      const v8bf lo = *(const v8bf*)(p + aOff);
      const v8bf hi = *(const v8bf*)(p + 16 + aOff);
      aF[s2] = __builtin_shufflevector(lo, hi, 0, 1, 2, 3, 4, 5, 6, 7,
                                               8, 9, 10, 11, 12, 13, 14, 15);
    }
#pragma unroll
    for (int t2 = 0; t2 < 4; ++t2) {
      const bf16* p = &Bs[buf][(wn * 64 + t2 * 16 + lrow) * LDST];
      const v8bf lo = *(const v8bf*)(p + bOff);
      const v8bf hi = *(const v8bf*)(p + 16 + bOff);
      bF[t2] = __builtin_shufflevector(lo, hi, 0, 1, 2, 3, 4, 5, 6, 7,
                                               8, 9, 10, 11, 12, 13, 14, 15);
    }
#pragma unroll
    for (int s2 = 0; s2 < 2; ++s2)
#pragma unroll
      for (int t2 = 0; t2 < 4; ++t2)
        acc[s2][t2] = __builtin_amdgcn_wmma_f32_16x16x32_bf16(
            false, aF[s2], false, bF[t2], (short)0, acc[s2][t2], false, false);
  };

  // --- 3-buffer pipeline: tiles T_i at k = 32*i.  Prologue fills T0, T1. ---
  issueTile(0, 0);
  issueTile(32, 1);

  int cur = 0;
  int k0 = 0;
  // main: iterations that also prefetch T_{i+2} (branch-free body)
  for (; k0 + 64 < K; k0 += 32) {
    wait_async_le4();            // T_i complete (in-order; only T_{i+1} after it)
    __syncthreads();             // all waves' copies into buffer 'cur' visible
    computeTile(cur);
    issueTile(k0 + 64, (cur == 0) ? 2 : cur - 1);  // refill buffer read last iter
    cur = (cur == 2) ? 0 : cur + 1;
  }
  // tail: second-to-last tile (T_{L-1} still outstanding)
  wait_async_le4();
  __syncthreads();
  computeTile(cur);
  cur = (cur == 2) ? 0 : cur + 1;
  // tail: last tile
  wait_async_0();
  __syncthreads();
  computeTile(cur);

  // --- epilogue: scale by gamma_w (+ optional exact GELU -> bf16) ---
  const float gw = *gammaPtr;
#pragma unroll
  for (int s2 = 0; s2 < 2; ++s2) {
#pragma unroll
    for (int t2 = 0; t2 < 4; ++t2) {
#pragma unroll
      for (int r = 0; r < 8; ++r) {
        const int m = mBase + wm * 32 + s2 * 16 + half * 8 + r;
        const int n = nBase + wn * 64 + t2 * 16 + lrow;
        const float v = acc[s2][t2][r] * gw;
        if (GELU_BF16) {
          const float ge = 0.5f * v * (1.0f + erff(v * 0.70710678118654752f));
          ((bf16*)Out)[(size_t)m * N + n] = (bf16)ge;
        } else {
          ((float*)Out)[(size_t)m * N + n] = v;
        }
      }
    }
  }
}

// ---------------------------------------------------------------------------
// Launch
// ---------------------------------------------------------------------------
extern "C" void kernel_launch(void* const* d_in, const int* in_sizes, int n_in,
                              void* d_out, int out_size, void* d_ws, size_t ws_size,
                              hipStream_t stream) {
  (void)in_sizes; (void)n_in; (void)out_size; (void)ws_size;
  const float* x      = (const float*)d_in[0];
  const float* w_fc   = (const float*)d_in[1];
  const float* w_proj = (const float*)d_in[2];
  float* out = (float*)d_out;

  char* ws = (char*)d_ws;
  bf16*  A1    = (bf16*)(ws + OFF_A1);
  bf16*  WfcQ  = (bf16*)(ws + OFF_WFCQ);
  bf16*  WprQ  = (bf16*)(ws + OFF_WPRQ);
  bf16*  Hbuf  = (bf16*)(ws + OFF_H);
  float* partA = (float*)(ws + OFF_PARTA);
  float* partB = (float*)(ws + OFF_PARTB);
  float* gFc   = (float*)(ws + OFF_GAMMA);
  float* gPr   = gFc + 1;

  const float invW = 1.0f / (float)WCNT;

  // 1) weight gammas (deterministic 2-stage reductions)
  absmean_stage1<<<2048, 256, 0, stream>>>(w_fc, partA);
  absmean_stage2<<<1, 256, 0, stream>>>(partA, gFc, invW);
  absmean_stage1<<<2048, 256, 0, stream>>>(w_proj, partB);
  absmean_stage2<<<1, 256, 0, stream>>>(partB, gPr, invW);

  // 2) ternarize weights to bf16 {-1,0,1}
  ternarize_k<<<WCNT / (256 * 4), 256, 0, stream>>>(w_fc, gFc, WfcQ, (size_t)WCNT);
  ternarize_k<<<WCNT / (256 * 4), 256, 0, stream>>>(w_proj, gPr, WprQ, (size_t)WCNT);

  // 3) LN + absmax clip of x -> bf16 A1
  ln_quant_x_k<<<MM, 256, 0, stream>>>(x, A1);

  // 4) GEMM1: (8192x2048) @ (2048x8192) -> GELU -> bf16 H
  gemm_wmma_k<true><<<dim3(HH / 128, MM / 128), 256, 0, stream>>>(
      A1, WfcQ, gFc, Hbuf, MM, HH, DD);

  // 5) LN + absmax clip of H (in place, bf16)
  ln_quant_h_k<<<MM, 256, 0, stream>>>(Hbuf);

  // 6) GEMM2: (8192x8192) @ (8192x2048) -> fp32 out
  gemm_wmma_k<false><<<dim3(DD / 128, MM / 128), 256, 0, stream>>>(
      Hbuf, WprQ, gPr, out, MM, DD, HH);
}
